// VanillaRNN_31439160607389
// MI455X (gfx1250) — matrix operand order repaired and checked by reading, not compiled, not run on hardware
//
#include <hip/hip_runtime.h>
#include <hip/hip_bf16.h>
#include <math.h>

// Vanilla RNN, fused over all 512 timesteps.
// Each workgroup: 32 batch rows, 512 threads = 16 wave32s.
// Wave w computes hidden columns [16w,16w+16) for all 32 rows:
//   16 x v_wmma_f32_16x16x32_bf16 per timestep (2 M-tiles x 8 K-chunks),
//   sharing one in-register W_hh B-fragment set across both M-tiles.
// Recurrent state h carried in LDS (bf16); activation via native v_tanh_f32.

#define BATCH   16384
#define SEQ     512
#define NH      256
#define NC      10
#define TILE_B  32
#define TCHUNK  64
#define H_STRIDE 264    // 256 + 8 bf16 pad: row stride = 132 dwords -> rows hit distinct banks
#define BLOCK   512

typedef __bf16 bf16_t;
typedef __attribute__((ext_vector_type(16))) __bf16 v16bf;
typedef __attribute__((ext_vector_type(8)))  __bf16 v8bf;
typedef __attribute__((ext_vector_type(8)))  float  v8f;

__device__ __forceinline__ float fast_tanh(float s) {
#if __has_builtin(__builtin_amdgcn_tanhf)
    return __builtin_amdgcn_tanhf(s);   // native v_tanh_f32 (gfx1250 TRANS op)
#else
    return tanhf(s);
#endif
}

__device__ __forceinline__ v16bf load_a_frag(const bf16_t* p0) {
    v8bf lo = *(const v8bf*)(p0);        // 16B -> ds_load_b128
    v8bf hi = *(const v8bf*)(p0 + 16);   // 16B -> ds_load_b128
    return __builtin_shufflevector(lo, hi, 0,1,2,3,4,5,6,7,8,9,10,11,12,13,14,15);
}

__global__ __launch_bounds__(BLOCK)
void VanillaRNN_fused_wmma_kernel(const float* __restrict__ x,
                                  const float* __restrict__ Whx,
                                  const float* __restrict__ Whh,
                                  const float* __restrict__ Wph,
                                  const float* __restrict__ bh,
                                  const float* __restrict__ bp,
                                  float* __restrict__ out)
{
    __shared__ bf16_t h_lds[TILE_B * H_STRIDE];   // recurrent state, bf16 (~17KB)
    __shared__ float  x_lds[TILE_B * TCHUNK];     // staged input chunk (8KB)
    __shared__ float  hf_lds[TILE_B * NH];        // fp32 copy of final h (32KB)

    const int tid  = threadIdx.x;
    const int wave = tid >> 5;
    const int lane = tid & 31;
    const int half = lane >> 4;     // 0: lanes 0-15, 1: lanes 16-31
    const int l15  = lane & 15;
    const int n0   = wave * 16;     // this wave's output-column tile
    const int b0   = blockIdx.x * TILE_B;

    // ---- Preload W_hh column slice [256 x 16] into registers as bf16 B-fragments.
    // B (KxN=32x16) layout: N = l15; e<8 -> K = half*8 + e, e>=8 -> K = 16 + half*8 + (e-8).
    v16bf bfrag[8];
    #pragma unroll
    for (int kc = 0; kc < 8; ++kc) {
        #pragma unroll
        for (int e = 0; e < 16; ++e) {
            int K = kc * 32 + half * 8 + (e & 7) + (e >> 3) * 16;
            bfrag[kc][e] = (bf16_t)Whh[K * NH + n0 + l15];
        }
    }
    const float whx_n = Whx[n0 + l15];   // INPUT_DIM == 1 -> rank-1 input projection
    const float bh_n  = bh[n0 + l15];

    // h0 = 0
    for (int i = tid; i < TILE_B * H_STRIDE; i += BLOCK) h_lds[i] = (bf16_t)0.0f;

    for (int tc = 0; tc < SEQ; tc += TCHUNK) {
        // stage x[b0..b0+31][tc..tc+63] (coalesced: 64 consecutive floats per row)
        for (int i = tid; i < TILE_B * TCHUNK; i += BLOCK) {
            int r = i >> 6;            // TCHUNK == 64
            int c = i & 63;
            x_lds[i] = x[(size_t)(b0 + r) * SEQ + tc + c];
        }
        if (tc + TCHUNK < SEQ) {
            __builtin_prefetch(&x[(size_t)b0 * SEQ + tc + TCHUNK], 0, 1);
        }
        __syncthreads();   // x chunk + (iter 0) h init visible

        for (int tl = 0; tl < TCHUNK; ++tl) {
            // Seed accumulators with b_h + x_t * W_hx.
            // C/D layout: VGPR v -> M = v + 8*half (tile A), M = 16 + v + 8*half (tile B), N = l15.
            v8f accA0, accA1, accB0, accB1;
            #pragma unroll
            for (int v = 0; v < 8; ++v) {
                int m = v + half * 8;
                accA0[v] = bh_n + x_lds[m * TCHUNK + tl] * whx_n;
                accB0[v] = bh_n + x_lds[(16 + m) * TCHUNK + tl] * whx_n;
                accA1[v] = 0.0f;
                accB1[v] = 0.0f;
            }

            // h @ W_hh : K = 256 as 8 chunks of 32; 2 M-tiles share each B fragment.
            // A (MxK=16x32) layout: M = l15; e<8 -> K = kc*32 + half*8 + e,
            //                                e>=8 -> K = kc*32 + 16 + half*8 + (e-8).
            #pragma unroll
            for (int kc = 0; kc < 8; ++kc) {
                const int koff = kc * 32 + half * 8;
                v16bf aA = load_a_frag(&h_lds[l15 * H_STRIDE + koff]);
                v16bf aB = load_a_frag(&h_lds[(16 + l15) * H_STRIDE + koff]);
                if (kc & 1) {
                    accA1 = __builtin_amdgcn_wmma_f32_16x16x32_bf16(
                                false, aA, false, bfrag[kc], (short)0, accA1, false, false);
                    accB1 = __builtin_amdgcn_wmma_f32_16x16x32_bf16(
                                false, aB, false, bfrag[kc], (short)0, accB1, false, false);
                } else {
                    accA0 = __builtin_amdgcn_wmma_f32_16x16x32_bf16(
                                false, aA, false, bfrag[kc], (short)0, accA0, false, false);
                    accB0 = __builtin_amdgcn_wmma_f32_16x16x32_bf16(
                                false, aB, false, bfrag[kc], (short)0, accB0, false, false);
                }
            }

            __syncthreads();   // every wave finished reading h

            const bool last = (tc + tl == SEQ - 1);
            #pragma unroll
            for (int v = 0; v < 8; ++v) {
                int m = v + half * 8;
                float hA = fast_tanh(accA0[v] + accA1[v]);
                float hB = fast_tanh(accB0[v] + accB1[v]);
                h_lds[m * H_STRIDE + n0 + l15]        = (bf16_t)hA;
                h_lds[(16 + m) * H_STRIDE + n0 + l15] = (bf16_t)hB;
                if (last) {
                    hf_lds[m * NH + n0 + l15]        = hA;
                    hf_lds[(16 + m) * NH + n0 + l15] = hB;
                }
            }
            __syncthreads();   // new h visible to all waves
        }
    }

    // Final projection: p = h_last @ W_ph + b_p  (32x10 per block -> trivial VALU)
    if (tid < TILE_B * NC) {
        int r = tid / NC, c = tid % NC;
        float s = bp[c];
        #pragma unroll 4
        for (int k = 0; k < NH; ++k)
            s += hf_lds[r * NH + k] * Wph[k * NC + c];
        out[(size_t)(b0 + r) * NC + c] = s;
    }
}

extern "C" void kernel_launch(void* const* d_in, const int* in_sizes, int n_in,
                              void* d_out, int out_size, void* d_ws, size_t ws_size,
                              hipStream_t stream) {
    (void)in_sizes; (void)n_in; (void)d_ws; (void)ws_size; (void)out_size;
    const float* x   = (const float*)d_in[0];
    const float* Whx = (const float*)d_in[1];
    const float* Whh = (const float*)d_in[2];
    const float* Wph = (const float*)d_in[3];
    const float* bh  = (const float*)d_in[4];
    const float* bp  = (const float*)d_in[5];
    float* out = (float*)d_out;

    dim3 grid(BATCH / TILE_B);   // 512 workgroups
    dim3 block(BLOCK);           // 16 wave32s
    hipLaunchKernelGGL(VanillaRNN_fused_wmma_kernel, grid, block, 0, stream,
                       x, Whx, Whh, Wph, bh, bp, out);
}
